// ChEBIRecNN_69827578298883
// MI455X (gfx1250) — compile-verified, matrix operand order checked by reading, not run on hardware
//
#include <hip/hip_runtime.h>
#include <hip/hip_bf16.h>
#include <math.h>

typedef __bf16 v16bf __attribute__((ext_vector_type(16)));
typedef float  v8f   __attribute__((ext_vector_type(8)));

#define L_FEAT 104
#define S_OUT  500
#define M_MOL  64
#define T_LVL  32
#define W_NODE 256
#define SEGPAD 128          // each 104-wide input segment padded to 128
#define NPAD   112          // 104 -> 7 tiles of 16
#define KPAD   384          // 3 segments * 128
#define NTHREADS 512        // 16 waves of 32

// LDS layout (bytes)
#define OFF_H    0                       // 256*128 bf16 = 65536
#define OFF_CTX  65536                   // 256*128 bf16 = 65536
#define OFF_WT   131072                  // 112*384 bf16 = 86016
#define OFF_BS   217088                  // 112 f32 (b_single padded)
#define OFF_BI   217600                  // 112 f32 (b_int2 padded)
#define SMEM_BYTES 218112

// One level: D = relu(A * Wt + bias), A rows are [ctx | h[p0] | h[p1]] (each
// segment padded to 128), Wt is transposed-padded weights [NPAD][KPAD] bf16.
// Each wave owns one 16-row M tile and all 7 N tiles.
__device__ __forceinline__ void level_gemm(
    __bf16* __restrict__ H, const __bf16* __restrict__ CTX,
    const __bf16* __restrict__ WT, const float* __restrict__ bias,
    int tile_m, int lane, int p0, int p1, int nkstep)
{
    v8f acc[7];
#pragma unroll
    for (int n = 0; n < 7; ++n)
#pragma unroll
        for (int v = 0; v < 8; ++v) acc[n][v] = 0.0f;

    const int hi16 = ((lane >> 4) & 1) << 4;   // lanes 16..31 take K+16 half
    const int nlo  = lane & 15;

    // unroll capped at 2: 56 accum VGPRs + 2 x (8 A + 56 B) in-flight frags
    // stays under the 256 directly-addressable VGPR limit (no spills).
#pragma unroll 2
    for (int s = 0; s < nkstep; ++s) {
        const int seg = s >> 2;                 // which 128-wide segment
        const int k0  = ((s & 3) << 5) + hi16;  // K offset within segment
        const __bf16* abase;
        if (seg == 0) {
            abase = CTX + (tile_m * 16 + nlo) * SEGPAD + k0;   // context seg
        } else {
            const int prow = (seg == 1) ? p0 : p1;             // parent rows
            abase = H + prow * SEGPAD + k0;
        }

        // Batch all fragment loads first so the scheduler can keep 16
        // ds_load_b128 in flight and count dscnt down instead of issuing
        // load->wait(0)->wmma chains.
        const int kw = (s << 5) + hi16;         // K offset into Wt rows
        v16bf bfrag[7];
#pragma unroll
        for (int n = 0; n < 7; ++n)
            bfrag[n] = *(const v16bf*)(WT + (n * 16 + nlo) * KPAD + kw);
        v16bf a = *(const v16bf*)abase;         // A frag: 16 bf16 (contiguous k = 16h + s)

#pragma unroll
        for (int n = 0; n < 7; ++n)
            acc[n] = __builtin_amdgcn_wmma_f32_16x16x32_bf16(
                false, a, false, bfrag[n], (short)0, acc[n], false, false);
    }

    __syncthreads();   // all reads of H (parent gathers) complete

    // bias + relu + bf16 store back into H (single buffer is safe post-barrier)
    const int rbase = tile_m * 16 + ((lane >> 4) << 3);
#pragma unroll
    for (int n = 0; n < 7; ++n) {
        const int col = n * 16 + nlo;           // padded cols 104..111 get 0
        const float bv = bias[col];
#pragma unroll
        for (int v = 0; v < 8; ++v) {
            float x = acc[n][v] + bv;
            x = x > 0.0f ? x : 0.0f;
            H[(rbase + v) * SEGPAD + col] = (__bf16)x;
        }
    }
    __syncthreads();
}

__global__ __launch_bounds__(NTHREADS, 1)
void dag_levels_kernel(const float* __restrict__ ctx_g,
                       const int*   __restrict__ pidx_g,
                       const int*   __restrict__ spidx_g,
                       const float* __restrict__ Wsg, const float* __restrict__ bsg,
                       const float* __restrict__ Wi,  const float* __restrict__ bi,
                       const float* __restrict__ Wsk, const float* __restrict__ bsk,
                       float* __restrict__ sink_ws)
{
    extern __shared__ __align__(128) char smem_raw[];
    __bf16* H   = (__bf16*)(smem_raw + OFF_H);
    __bf16* CTX = (__bf16*)(smem_raw + OFF_CTX);
    __bf16* WT  = (__bf16*)(smem_raw + OFF_WT);
    float*  BS  = (float*)(smem_raw + OFF_BS);
    float*  BI  = (float*)(smem_raw + OFF_BI);

    const int m      = blockIdx.x;
    const int tid    = threadIdx.x;
    const int lane   = tid & 31;
    const int tile_m = tid >> 5;           // wave id = M tile
    const int nlo    = lane & 15;

    // --- one-time init: zero padded LDS regions, stage biases ---
    for (int i = tid; i < W_NODE * SEGPAD; i += NTHREADS) {
        H[i]   = (__bf16)0.0f;
        CTX[i] = (__bf16)0.0f;
    }
    if (tid < NPAD) {
        BS[tid] = (tid < L_FEAT) ? bsg[tid] : 0.0f;
        BI[tid] = (tid < L_FEAT) ? bi[tid]  : 0.0f;
    }
    // stage transposed-padded W_single into WT (rows stride KPAD, only seg 0)
    for (int i = tid; i < NPAD * SEGPAD; i += NTHREADS) {
        const int n = i >> 7, kp = i & 127;
        float v = (n < L_FEAT && kp < L_FEAT) ? Wsg[kp * L_FEAT + n] : 0.0f;
        WT[n * KPAD + kp] = (__bf16)v;
    }
    // stage context level 0
    {
        const float* src = ctx_g + ((size_t)m * T_LVL + 0) * (W_NODE * L_FEAT);
        for (int i = tid; i < W_NODE * L_FEAT; i += NTHREADS) {
            const int w = i / L_FEAT, l = i - w * L_FEAT;
            CTX[w * SEGPAD + l] = (__bf16)src[i];
        }
    }
    __syncthreads();

    // --- level 0: h0 = relu(ctx0 @ W_single + b_single), K = 1 segment ---
    level_gemm(H, CTX, WT, BS, tile_m, lane, 0, 0, 4);

    // stage transposed-padded W_int2 [NPAD][KPAD] (3 segments of 104 -> 128)
    for (int i = tid; i < NPAD * KPAD; i += NTHREADS) {
        const int n = i / KPAD, kp = i - n * KPAD;
        const int seg = kp >> 7, kin = kp & 127;
        float v = 0.0f;
        if (n < L_FEAT && kin < L_FEAT)
            v = Wi[(seg * L_FEAT + kin) * L_FEAT + n];
        WT[n * KPAD + kp] = (__bf16)v;
    }

    // --- levels 1..31 ---
    for (int t = 1; t < T_LVL; ++t) {
        // stage context level t (previous level's CTX reads finished in gemm)
        const float* src = ctx_g + ((size_t)m * T_LVL + t) * (W_NODE * L_FEAT);
        for (int i = tid; i < W_NODE * L_FEAT; i += NTHREADS) {
            const int w = i / L_FEAT, l = i - w * L_FEAT;
            CTX[w * SEGPAD + l] = (__bf16)src[i];
        }
        // prefetch next level's context so HBM latency overlaps this level's
        // WMMA work (lowers to global_prefetch_b8 on gfx1250)
        if (t + 1 < T_LVL) {
            const char* nsrc = (const char*)(ctx_g +
                ((size_t)m * T_LVL + t + 1) * (W_NODE * L_FEAT));
            // 256*104*4 B = 832 cachelines of 128 B; 512 threads -> 2 each
            __builtin_prefetch(nsrc + (size_t)tid * 256, 0, 3);
            __builtin_prefetch(nsrc + (size_t)tid * 256 + 128, 0, 3);
        }
        // parent indices for this lane's row (lanes 16..31 mirror rows 0..15)
        const int r = tile_m * 16 + nlo;
        const size_t pbase = (((size_t)m * (T_LVL - 1) + (t - 1)) * W_NODE + r) * 2;
        const int p0 = pidx_g[pbase + 0];
        const int p1 = pidx_g[pbase + 1];
        __syncthreads();   // CTX staged (and WT, on first iteration)

        level_gemm(H, CTX, WT, BI, tile_m, lane, p0, p1, 12);
    }

    // --- sink: concat 7 parent rows -> Linear(728, 500), per molecule ---
    if (tid < S_OUT) {
        float acc = bsk[tid];
        for (int j = 0; j < 7; ++j) {
            const int pr = spidx_g[m * 7 + j];
            const float* wc = Wsk + (size_t)(j * L_FEAT) * S_OUT + tid;
            for (int k = 0; k < L_FEAT; ++k)
                acc += (float)H[pr * SEGPAD + k] * wc[(size_t)k * S_OUT];
        }
        sink_ws[m * S_OUT + tid] = acc;
    }
}

__global__ __launch_bounds__(NTHREADS)
void finalize_kernel(const float* __restrict__ sink_ws,
                     const float* __restrict__ Wcls,
                     const float* __restrict__ bcls,
                     float* __restrict__ out)
{
    __shared__ float avg[S_OUT];
    const int tid = threadIdx.x;
    if (tid < S_OUT) {
        float s = 0.0f;
        for (int mm = 0; mm < M_MOL; ++mm) s += sink_ws[mm * S_OUT + tid];
        avg[tid] = s * (1.0f / (float)M_MOL);
    }
    __syncthreads();
    if (tid < S_OUT) {
        float acc = bcls[tid];
        for (int k = 0; k < S_OUT; ++k)
            acc += avg[k] * Wcls[(size_t)k * S_OUT + tid];
        out[tid] = 1.0f / (1.0f + expf(-acc));
    }
}

extern "C" void kernel_launch(void* const* d_in, const int* in_sizes, int n_in,
                              void* d_out, int out_size, void* d_ws, size_t ws_size,
                              hipStream_t stream) {
    const float* ctx   = (const float*)d_in[0];
    const int*   pidx  = (const int*)  d_in[1];
    const int*   spidx = (const int*)  d_in[2];
    const float* Wsg   = (const float*)d_in[3];
    const float* bsg   = (const float*)d_in[4];
    const float* Wi    = (const float*)d_in[5];
    const float* bi    = (const float*)d_in[6];
    const float* Wsk   = (const float*)d_in[7];
    const float* bsk   = (const float*)d_in[8];
    const float* Wcls  = (const float*)d_in[9];
    const float* bcls  = (const float*)d_in[10];
    float* out = (float*)d_out;
    float* ws  = (float*)d_ws;   // 64*500 f32 = 128000 B

    (void)in_sizes; (void)n_in; (void)out_size; (void)ws_size;

    (void)hipFuncSetAttribute((const void*)dag_levels_kernel,
                              hipFuncAttributeMaxDynamicSharedMemorySize,
                              SMEM_BYTES);

    dag_levels_kernel<<<M_MOL, NTHREADS, SMEM_BYTES, stream>>>(
        ctx, pidx, spidx, Wsg, bsg, Wi, bi, Wsk, bsk, ws);
    finalize_kernel<<<1, NTHREADS, 0, stream>>>(ws, Wcls, bcls, out);
}
